// GPT2Block_3375844295040
// MI455X (gfx1250) — matrix operand order repaired and checked by reading, not compiled, not run on hardware
//
#include <hip/hip_runtime.h>
#include <hip/hip_bf16.h>
#include <math.h>

// ---------------------------------------------------------------------------
// GPT-2 block forward for gfx1250 (MI455X).  All GEMM contractions via
// v_wmma_f32_16x16x32_bf16 (wave32).  GEMM operand staging via the Tensor
// Data Mover (tensor_load_to_lds, double-buffered, TENSORcnt-tracked).
// fp32 in/out, bf16 internal.
// ---------------------------------------------------------------------------

typedef __bf16 bf16_t;
typedef __attribute__((ext_vector_type(16))) __bf16 v16bf;
typedef __attribute__((ext_vector_type(8)))  __bf16 v8bf;
typedef __attribute__((ext_vector_type(8)))  float  v8f;
typedef __attribute__((ext_vector_type(4)))  unsigned int v4u;
typedef __attribute__((ext_vector_type(4)))  int v4i;
typedef __attribute__((ext_vector_type(8)))  int v8i;

#define D_EMBD 1024
#define N_HEAD 16
#define HEAD_D 64
#define T_SEQ  2048
#define N_BATCH 2
#define M_ROWS (N_BATCH * T_SEQ)   // 4096

static __device__ inline v8f zero8() {
  v8f z;
#pragma unroll
  for (int i = 0; i < 8; ++i) z[i] = 0.0f;
  return z;
}

// A-operand (16x32 bf16, M x K): lane m = lane%16; lanes 0-15 hold K {0..15},
// lanes 16-31 hold K {16..31}.  `row` points at A[m][kb].
static __device__ inline v16bf load_a_frag(const bf16_t* row, int half) {
  const bf16_t* p = row + (half ? 16 : 0);
  v8bf lo = *(const v8bf*)(p);
  v8bf hi = *(const v8bf*)(p + 8);
  v16bf r;
#pragma unroll
  for (int i = 0; i < 8; ++i) { r[i] = lo[i]; r[i + 8] = hi[i]; }
  return r;
}

static __device__ inline v8f wmma_bf16(v16bf a, v16bf b, v8f c) {
  return __builtin_amdgcn_wmma_f32_16x16x32_bf16(
      /*neg_a=*/false, a, /*neg_b=*/false, b,
      /*c_mod=*/(short)0, c, /*reuse_a=*/false, /*reuse_b=*/false);
}

static __device__ inline float gelu_tanh(float x) {
  float x3 = x * x * x;
  float t = tanhf(0.79788456080286535588f * (x + 0.044715f * x3));
  return 0.5f * x * (1.0f + t);
}

// ---------------------------------------------------------------------------
// TDM 2D tile load: global (row-major, bf16, row stride = strideElems) -> LDS
// (tile rows stored contiguously, tile_k*2 bytes per row).  D# packed per
// CDNA5 ISA 8.3/8.4 (group0: count/lds/global/type=2; group1: data_size=2B,
// tensor dims, tile dims, dim0 stride).  Groups 2/3 zero (2D tensor).
// This toolchain exposes the 6-arg builtin: (v4u, v8i, v4i, v4i, v8i, i32).
// ---------------------------------------------------------------------------
static __device__ inline void tdm_load_tile_2d(unsigned lds_byte_off,
                                               const bf16_t* gptr,
                                               unsigned tile_k, unsigned tile_rows,
                                               unsigned tensor_k, unsigned tensor_rows,
                                               unsigned long long stride_elems) {
  const unsigned long long ga = (unsigned long long)(uintptr_t)gptr;
  v4u g0;
  g0.x = 1u;                                                   // count=1 (user D#)
  g0.y = lds_byte_off;                                         // lds_addr (bytes)
  g0.z = (unsigned)ga;                                         // global_addr[31:0]
  g0.w = (unsigned)((ga >> 32) & 0x01ffffffu) | (2u << 30);    // addr[56:32] | type=2
  v8i g1;
  g1[0] = (int)(1u << 16);                                     // data_size=1 -> 2 bytes
  g1[1] = (int)((tensor_k & 0xffffu) << 16);                   // tensor_dim0[15:0]
  g1[2] = (int)((tensor_k >> 16) | ((tensor_rows & 0xffffu) << 16));
  g1[3] = (int)((tensor_rows >> 16) | ((tile_k & 0xffffu) << 16));  // tile_dim0
  g1[4] = (int)(tile_rows & 0xffffu);                          // tile_dim1 (tile_dim2=0)
  g1[5] = (int)(unsigned)(stride_elems & 0xffffffffu);         // dim0_stride[31:0]
  g1[6] = (int)(unsigned)((stride_elems >> 32) & 0xffffu);     // dim0_stride[47:32]
  g1[7] = 0;                                                   // dim1_stride (unused, 2D)
  const v4i gz4 = {0, 0, 0, 0};
  const v8i gz8 = {0, 0, 0, 0, 0, 0, 0, 0};
  __builtin_amdgcn_tensor_load_to_lds(g0, g1, gz4, gz4, gz8, 0);
}

// ---------------------------------------------------------------------------
// Weight cast + transpose: W[K][N] f32 (row-major) -> Wt[N][K] bf16.
// ---------------------------------------------------------------------------
__global__ __launch_bounds__(256) void castT_kernel(const float* __restrict__ W,
                                                    bf16_t* __restrict__ Wt,
                                                    int K, int N) {
  __shared__ float tile[32][33];
  int kb = blockIdx.y * 32, nb = blockIdx.x * 32;
  int tx = threadIdx.x, ty = threadIdx.y;  // 32 x 8
#pragma unroll
  for (int i = 0; i < 32; i += 8)
    tile[ty + i][tx] = W[(size_t)(kb + ty + i) * N + nb + tx];
  __syncthreads();
#pragma unroll
  for (int i = 0; i < 32; i += 8)
    Wt[(size_t)(nb + ty + i) * K + kb + tx] = (bf16_t)tile[tx][ty + i];
}

// ---------------------------------------------------------------------------
// LayerNorm over last dim (D=1024) + cast to bf16.  One block per row.
// ---------------------------------------------------------------------------
__global__ __launch_bounds__(256) void ln_cast_kernel(const float* __restrict__ x,
                                                      const float* __restrict__ g,
                                                      const float* __restrict__ bt,
                                                      bf16_t* __restrict__ out) {
  __shared__ float red[256];
  __shared__ float stat[2];
  const int tid = threadIdx.x;
  const size_t row = blockIdx.x;
  const float4 v = ((const float4*)(x + row * D_EMBD))[tid];
  red[tid] = v.x + v.y + v.z + v.w;
  __syncthreads();
  for (int o = 128; o > 0; o >>= 1) {
    if (tid < o) red[tid] += red[tid + o];
    __syncthreads();
  }
  if (tid == 0) stat[0] = red[0] * (1.0f / D_EMBD);
  __syncthreads();
  const float mean = stat[0];
  float dx = v.x - mean, dy = v.y - mean, dz = v.z - mean, dw = v.w - mean;
  red[tid] = dx * dx + dy * dy + dz * dz + dw * dw;
  __syncthreads();
  for (int o = 128; o > 0; o >>= 1) {
    if (tid < o) red[tid] += red[tid + o];
    __syncthreads();
  }
  if (tid == 0) stat[1] = rsqrtf(red[0] * (1.0f / D_EMBD) + 1e-5f);
  __syncthreads();
  const float rstd = stat[1];
  const int c = tid * 4;
  const float4 gw = ((const float4*)g)[tid];
  const float4 bb = ((const float4*)bt)[tid];
  bf16_t* o = out + row * D_EMBD + c;
  o[0] = (bf16_t)(dx * rstd * gw.x + bb.x);
  o[1] = (bf16_t)(dy * rstd * gw.y + bb.y);
  o[2] = (bf16_t)(dz * rstd * gw.z + bb.z);
  o[3] = (bf16_t)(dw * rstd * gw.w + bb.w);
}

// ---------------------------------------------------------------------------
// WMMA GEMM with TDM-staged, double-buffered LDS tiles:
//   out[M][N] = A[M][K](bf16) * Wt[N][K]^T(bf16) + bias  (+resid, +GELU)
// Block: 256 thr = 8 waves -> 128(M) x 64(N) tile; K chunked by 64.
// Wave 0 issues tensor_load_to_lds for the next chunk (A 128x64, B 64x64)
// while all 8 waves run WMMA out of the current LDS buffer.
// ---------------------------------------------------------------------------
template <bool GELU_ACT, bool OUT_BF, bool RES>
__global__ __launch_bounds__(256) void gemm_wmma_kernel(
    const bf16_t* __restrict__ A, const bf16_t* __restrict__ Wt,
    const float* __restrict__ bias, const float* __restrict__ resid,
    float* __restrict__ outF, bf16_t* __restrict__ outB,
    int M, int N, int K) {
  __shared__ __align__(16) bf16_t Ald[2][128][64];   // 32 KB
  __shared__ __align__(16) bf16_t Bld[2][64][64];    // 16 KB
  const int wave = threadIdx.x >> 5, lane = threadIdx.x & 31;
  const int lm = lane & 15, lh = lane >> 4;
  const int mBlk = blockIdx.y * 128;
  const int n0 = blockIdx.x * 64;

  v8f acc[4];
#pragma unroll
  for (int j = 0; j < 4; ++j) acc[j] = zero8();

  const bf16_t* Abase = A + (size_t)mBlk * K;
  const bf16_t* Bbase = Wt + (size_t)n0 * K;

  if (wave == 0) {
    tdm_load_tile_2d((unsigned)(uintptr_t)&Ald[0][0][0], Abase, 64, 128,
                     (unsigned)K, 128, (unsigned long long)K);
    tdm_load_tile_2d((unsigned)(uintptr_t)&Bld[0][0][0], Bbase, 64, 64,
                     (unsigned)K, 64, (unsigned long long)K);
  }
  __builtin_amdgcn_s_wait_tensorcnt(0);
  __syncthreads();

  int p = 0;
  for (int kb = 0; kb < K; kb += 64) {
    if (wave == 0 && kb + 64 < K) {   // prefetch next chunk into back buffer
      tdm_load_tile_2d((unsigned)(uintptr_t)&Ald[p ^ 1][0][0], Abase + kb + 64,
                       64, 128, (unsigned)(K - kb - 64), 128, (unsigned long long)K);
      tdm_load_tile_2d((unsigned)(uintptr_t)&Bld[p ^ 1][0][0], Bbase + kb + 64,
                       64, 64, (unsigned)(K - kb - 64), 64, (unsigned long long)K);
    }
#pragma unroll
    for (int ks = 0; ks < 64; ks += 32) {
      v16bf a = load_a_frag(&Ald[p][wave * 16 + lm][ks], lh);
#pragma unroll
      for (int j = 0; j < 4; ++j) {
        v16bf bm = *(const v16bf*)(&Bld[p][j * 16 + lm][ks + lh * 16]);
        acc[j] = wmma_bf16(a, bm, acc[j]);
      }
    }
    __builtin_amdgcn_s_wait_tensorcnt(0);  // TDM for next chunk complete
    __syncthreads();                        // all waves done reading buf p
    p ^= 1;
  }

  const int m0 = mBlk + wave * 16;
#pragma unroll
  for (int j = 0; j < 4; ++j) {
    const int n = n0 + j * 16 + lm;
    const float bv = bias[n];
#pragma unroll
    for (int r = 0; r < 8; ++r) {
      const int m = m0 + r + lh * 8;
      float v = acc[j][r] + bv;
      if (RES) v += resid[(size_t)m * N + n];
      if (GELU_ACT) v = gelu_tanh(v);
      if (OUT_BF) outB[(size_t)m * N + n] = (bf16_t)v;
      else        outF[(size_t)m * N + n] = v;
    }
  }
}

// ---------------------------------------------------------------------------
// Repack qkv f32 [B*T][3D] -> Qh[BH][T][64] (pre-scaled), Kh[BH][T][64],
// Vt[BH][64][T] (transposed for P@V B-operand), all bf16.
// ---------------------------------------------------------------------------
__global__ __launch_bounds__(256) void qkv_repack_kernel(const float* __restrict__ qkv,
                                                         bf16_t* __restrict__ Qh,
                                                         bf16_t* __restrict__ Kh,
                                                         bf16_t* __restrict__ Vt) {
  const int t = blockIdx.x;            // 0..B*T-1
  const int b = t / T_SEQ, ti = t % T_SEQ;
  const float* row = qkv + (size_t)t * (3 * D_EMBD);
  for (int c = threadIdx.x; c < D_EMBD; c += 256) {
    const int h = c >> 6, d = c & 63;
    const int bh = b * N_HEAD + h;
    const float q = row[c];
    const float k = row[D_EMBD + c];
    const float v = row[2 * D_EMBD + c];
    Qh[((size_t)bh * T_SEQ + ti) * HEAD_D + d] = (bf16_t)(q * 0.125f);  // 1/sqrt(64)
    Kh[((size_t)bh * T_SEQ + ti) * HEAD_D + d] = (bf16_t)k;
    Vt[((size_t)bh * HEAD_D + d) * T_SEQ + ti] = (bf16_t)v;
  }
}

// ---------------------------------------------------------------------------
// Causal flash attention, one wave per 16-row query tile, 32 keys per step.
// Block = 128 thr (4 waves); grid = (B*H, T/64).  EXEC stays all-1s for WMMA;
// per-wave LDS tiles ordered with s_wait_dscnt (DS in-order within a wave).
// ---------------------------------------------------------------------------
__global__ __launch_bounds__(128) void attn_kernel(const bf16_t* __restrict__ Qh,
                                                   const bf16_t* __restrict__ Kh,
                                                   const bf16_t* __restrict__ Vt,
                                                   bf16_t* __restrict__ ctx) {
  __shared__ float  Sld[4][16][33];
  __shared__ __align__(64) bf16_t Pld[4][16][32];
  __shared__ float rowmax[4][16], rowsum[4][16], scalef[4][16];

  const int wave = threadIdx.x >> 5, lane = threadIdx.x & 31;
  const int lm = lane & 15, lh = lane >> 4;
  const int bh = blockIdx.x;
  const int b = bh >> 4, h = bh & 15;
  const int m0 = blockIdx.y * 64 + wave * 16;

  const bf16_t* Q = Qh + (size_t)bh * T_SEQ * HEAD_D;
  const bf16_t* K = Kh + (size_t)bh * T_SEQ * HEAD_D;
  const bf16_t* V = Vt + (size_t)bh * HEAD_D * T_SEQ;

  const bf16_t* qrow = Q + (size_t)(m0 + lm) * HEAD_D;
  const v16bf aq0 = load_a_frag(qrow, lh);        // d 0..31
  const v16bf aq1 = load_a_frag(qrow + 32, lh);   // d 32..63

  v8f acc[4];
#pragma unroll
  for (int j = 0; j < 4; ++j) acc[j] = zero8();

  if (lane < 16) { rowmax[wave][lane] = -1e30f; rowsum[wave][lane] = 0.0f; }
  asm volatile("s_wait_dscnt 0x0" ::: "memory");

  for (int jb = 0; jb < m0 + 16; jb += 32) {
    // ---- S = Q K^T for 16 rows x 32 keys (two 16x16 WMMA tiles) ----
#pragma unroll
    for (int c = 0; c < 2; ++c) {
      const int j0 = jb + c * 16;
      const bf16_t* krow = K + (size_t)(j0 + lm) * HEAD_D + lh * 16;
      v16bf b0 = *(const v16bf*)(krow);        // d  0..31 slice
      v16bf b1 = *(const v16bf*)(krow + 32);   // d 32..63 slice
      v8f s = zero8();
      s = wmma_bf16(aq0, b0, s);
      s = wmma_bf16(aq1, b1, s);
#pragma unroll
      for (int r = 0; r < 8; ++r) {
        const int qi = m0 + r + lh * 8;
        const int kj = j0 + lm;
        Sld[wave][r + lh * 8][c * 16 + lm] = (kj <= qi) ? s[r] : -1e30f;
      }
    }
    asm volatile("s_wait_dscnt 0x0" ::: "memory");

    // ---- online softmax: lanes 0..15 each own one row ----
    if (lane < 16) {
      float mx = rowmax[wave][lane];
      float tm = mx;
#pragma unroll
      for (int j = 0; j < 32; ++j) tm = fmaxf(tm, Sld[wave][lane][j]);
      const float rs = __expf(mx - tm);
      float ps = 0.0f;
#pragma unroll
      for (int j = 0; j < 32; ++j) {
        const float e = __expf(Sld[wave][lane][j] - tm);
        ps += e;
        Pld[wave][lane][j] = (bf16_t)e;
      }
      rowmax[wave][lane] = tm;
      rowsum[wave][lane] = rowsum[wave][lane] * rs + ps;
      scalef[wave][lane] = rs;
    }
    asm volatile("s_wait_dscnt 0x0" ::: "memory");

    // ---- rescale accumulators, then O += P @ V ----
    float f[8];
#pragma unroll
    for (int r = 0; r < 8; ++r) f[r] = scalef[wave][r + lh * 8];
#pragma unroll
    for (int dt = 0; dt < 4; ++dt)
#pragma unroll
      for (int r = 0; r < 8; ++r) acc[dt][r] *= f[r];

    const v16bf ap = load_a_frag(&Pld[wave][lm][0], lh);   // P as A-operand (from LDS)
#pragma unroll
    for (int dt = 0; dt < 4; ++dt) {
      const bf16_t* vrow = V + (size_t)(dt * 16 + lm) * T_SEQ + jb + lh * 16;
      v16bf bv = *(const v16bf*)(vrow);
      acc[dt] = wmma_bf16(ap, bv, acc[dt]);
    }
  }

  asm volatile("s_wait_dscnt 0x0" ::: "memory");
  float inv[8];
#pragma unroll
  for (int r = 0; r < 8; ++r) inv[r] = 1.0f / rowsum[wave][r + lh * 8];
#pragma unroll
  for (int dt = 0; dt < 4; ++dt)
#pragma unroll
    for (int r = 0; r < 8; ++r) {
      const int qi = m0 + r + lh * 8;
      const float o = acc[dt][r] * inv[r];
      ctx[((size_t)(b * T_SEQ + qi)) * D_EMBD + h * HEAD_D + dt * 16 + lm] = (bf16_t)o;
    }
}

// ---------------------------------------------------------------------------
// Host-side orchestration
// ---------------------------------------------------------------------------
extern "C" void kernel_launch(void* const* d_in, const int* in_sizes, int n_in,
                              void* d_out, int out_size, void* d_ws, size_t ws_size,
                              hipStream_t stream) {
  (void)in_sizes; (void)n_in; (void)out_size; (void)ws_size;
  const float* x       = (const float*)d_in[0];
  const float* ln1_w   = (const float*)d_in[1];
  const float* ln1_b   = (const float*)d_in[2];
  const float* w_qkv   = (const float*)d_in[3];
  const float* b_qkv   = (const float*)d_in[4];
  const float* w_attn  = (const float*)d_in[5];
  const float* b_attn  = (const float*)d_in[6];
  const float* ln2_w   = (const float*)d_in[7];
  const float* ln2_b   = (const float*)d_in[8];
  const float* w_fc    = (const float*)d_in[9];
  const float* b_fc    = (const float*)d_in[10];
  const float* w_proj  = (const float*)d_in[11];
  const float* b_proj  = (const float*)d_in[12];
  float* out = (float*)d_out;

  const int D = D_EMBD, TD = 3 * D, FD = 4 * D, M = M_ROWS;

  size_t off = 0;
  char* base = (char*)d_ws;
  auto alloc = [&](size_t bytes) -> void* {
    void* p = base + off;
    off += (bytes + 255) & ~(size_t)255;
    return p;
  };
  bf16_t* wqkvT  = (bf16_t*)alloc((size_t)TD * D * 2);
  bf16_t* wattnT = (bf16_t*)alloc((size_t)D * D * 2);
  bf16_t* wfcT   = (bf16_t*)alloc((size_t)FD * D * 2);
  bf16_t* wprojT = (bf16_t*)alloc((size_t)D * FD * 2);
  bf16_t* h1     = (bf16_t*)alloc((size_t)M * D * 2);
  float*  qkvF   = (float*) alloc((size_t)M * TD * 4);
  bf16_t* Qh     = (bf16_t*)alloc((size_t)N_BATCH * N_HEAD * T_SEQ * HEAD_D * 2);
  bf16_t* Kh     = (bf16_t*)alloc((size_t)N_BATCH * N_HEAD * T_SEQ * HEAD_D * 2);
  bf16_t* Vt     = (bf16_t*)alloc((size_t)N_BATCH * N_HEAD * T_SEQ * HEAD_D * 2);
  bf16_t* ctx    = (bf16_t*)alloc((size_t)M * D * 2);
  float*  x2     = (float*) alloc((size_t)M * D * 4);
  bf16_t* h2     = (bf16_t*)alloc((size_t)M * D * 2);
  bf16_t* fcB    = (bf16_t*)alloc((size_t)M * FD * 2);

  const dim3 tB(32, 8);
  // Weight cast+transpose (grid = (N/32, K/32))
  castT_kernel<<<dim3(TD / 32, D / 32), tB, 0, stream>>>(w_qkv,  wqkvT,  D,  TD);
  castT_kernel<<<dim3(D  / 32, D / 32), tB, 0, stream>>>(w_attn, wattnT, D,  D);
  castT_kernel<<<dim3(FD / 32, D / 32), tB, 0, stream>>>(w_fc,   wfcT,   D,  FD);
  castT_kernel<<<dim3(D  / 32, FD / 32), tB, 0, stream>>>(w_proj, wprojT, FD, D);

  // LN1 -> h1 (bf16)
  ln_cast_kernel<<<M, 256, 0, stream>>>(x, ln1_w, ln1_b, h1);

  // qkv = h1 @ w_qkv + b_qkv  (f32)
  gemm_wmma_kernel<false, false, false><<<dim3(TD / 64, M / 128), 256, 0, stream>>>(
      h1, wqkvT, b_qkv, nullptr, qkvF, nullptr, M, TD, D);

  // split/scale/transpose heads
  qkv_repack_kernel<<<M, 256, 0, stream>>>(qkvF, Qh, Kh, Vt);

  // causal flash attention -> ctx (bf16, [B,T,D])
  attn_kernel<<<dim3(N_BATCH * N_HEAD, T_SEQ / 64), 128, 0, stream>>>(Qh, Kh, Vt, ctx);

  // x2 = x + ctx @ w_attn_proj + b_attn  (f32)
  gemm_wmma_kernel<false, false, true><<<dim3(D / 64, M / 128), 256, 0, stream>>>(
      ctx, wattnT, b_attn, x, x2, nullptr, M, D, D);

  // LN2 -> h2 (bf16)
  ln_cast_kernel<<<M, 256, 0, stream>>>(x2, ln2_w, ln2_b, h2);

  // fc = gelu(h2 @ w_fc + b_fc)  (bf16)
  gemm_wmma_kernel<true, true, false><<<dim3(FD / 64, M / 128), 256, 0, stream>>>(
      h2, wfcT, b_fc, nullptr, nullptr, fcB, M, FD, D);

  // out = x2 + fc @ w_proj + b_proj  (f32)
  gemm_wmma_kernel<false, false, true><<<dim3(D / 64, M / 128), 256, 0, stream>>>(
      fcB, wprojT, b_proj, x2, out, nullptr, M, D, FD);
}
